// MultiHeadAttention_10660108829383
// MI455X (gfx1250) — matrix operand order repaired and checked by reading, not compiled, not run on hardware
//
#include <hip/hip_runtime.h>
#include <hip/hip_bf16.h>
#include <math.h>
#include <stdint.h>

// ---------------------------------------------------------------------------
// Types for CDNA5 WMMA bf16 path
// ---------------------------------------------------------------------------
typedef __attribute__((ext_vector_type(16))) __bf16 bf16x16;
typedef __attribute__((ext_vector_type(8)))  __bf16 bf16x8;
typedef __attribute__((ext_vector_type(8)))  float  f32x8;
typedef __attribute__((ext_vector_type(4)))  unsigned int u32x4;
typedef __attribute__((ext_vector_type(8)))  int   i32x8;
typedef __attribute__((ext_vector_type(4)))  int   i32x4;

#define B_DIM 2
#define S_DIM 2048
#define C_DIM 1024
#define H_DIM 16
#define D_HEAD 64
#define W_EXP 4
#define M_ROWS (B_DIM * S_DIM)   // 4096

__device__ inline __bf16 to_bf16(float f) {
  union { float f; unsigned u; } v; v.f = f;
  unsigned r = v.u + 0x7FFFu + ((v.u >> 16) & 1u);   // round-to-nearest-even
  unsigned short h = (unsigned short)(r >> 16);
  __bf16 b; __builtin_memcpy(&b, &h, 2);
  return b;
}

__device__ inline float gelu_exact(float x) {
  return 0.5f * x * (1.0f + erff(x * 0.7071067811865475f));
}

// A fragment: 16x32 bf16 (MxK), row-major src with leading dim ld.
// Per ISA: lanes 0-15 m=lane, K 0..7 then 16..23; lanes 16-31: K 8..15 then 24..31.
__device__ inline bf16x16 load_a(const __bf16* base, int ld, int row0, int k0, int lane) {
  int m = lane & 15, hf = lane >> 4;
  const __bf16* p = base + (size_t)(row0 + m) * (size_t)ld + (size_t)(k0 + hf * 8);
  bf16x8 lo = *(const bf16x8*)p;
  bf16x8 hi = *(const bf16x8*)(p + 16);
  bf16x16 r;
#pragma unroll
  for (int i = 0; i < 8; ++i) { r[i] = lo[i]; r[i + 8] = hi[i]; }
  return r;
}

// B fragment: 32x16 bf16 (KxN) from transposed [N,K] storage with leading dim ld.
// Dense 16-bit B layout: lanes 0-15 hold K 0..7 then 16..23; lanes 16-31: K 8..15 then 24..31.
__device__ inline bf16x16 load_b(const __bf16* baseT, int ld, int n0, int k0, int lane) {
  int n = lane & 15, hf = lane >> 4;
  const __bf16* p = baseT + (size_t)(n0 + n) * (size_t)ld + (size_t)(k0 + hf * 8);
  bf16x8 lo = *(const bf16x8*)p;
  bf16x8 hi = *(const bf16x8*)(p + 16);
  bf16x16 r;
#pragma unroll
  for (int i = 0; i < 8; ++i) { r[i] = lo[i]; r[i + 8] = hi[i]; }
  return r;
}

__device__ inline f32x8 bf16_mma(bf16x16 a, bf16x16 b, f32x8 c) {
  return __builtin_amdgcn_wmma_f32_16x16x32_bf16(false, a, false, b, (short)0, c, false, false);
}

// ---------------------------------------------------------------------------
// Tensor Data Mover: 2D tile load Global -> LDS (bf16 elements, data_size=2B).
// D# packing per CDNA5 ISA ch.8 (group0 128b, group1 256b; groups 2/3 unused).
// Toolchain uses the 6-arg builtin: (u32x4, i32x8, i32x4, i32x4, i32x8, cpol).
// ---------------------------------------------------------------------------
__device__ inline void tdm_load_2d(unsigned lds_addr, const __bf16* gptr,
                                   unsigned tile_k, unsigned tile_rows,
                                   unsigned tensor_k, unsigned tensor_rows,
                                   unsigned stride_elems) {
  unsigned long long ga = (unsigned long long)(uintptr_t)gptr;
  u32x4 g0;
  g0[0] = 1u;                                    // count=1, user descriptor
  g0[1] = lds_addr;                              // LDS byte address
  g0[2] = (unsigned)ga;                          // global_addr[31:0]
  g0[3] = (unsigned)(ga >> 32) | (2u << 30);     // global_addr[56:32] | type=2
  i32x8 g1;
  g1[0] = (int)(1u << 16);                       // wg_mask=0, data_size=1 (2 bytes)
  g1[1] = (int)((tensor_k & 0xFFFFu) << 16);                     // tensor_dim0[15:0]
  g1[2] = (int)((tensor_k >> 16) | ((tensor_rows & 0xFFFFu) << 16)); // dim0 hi | dim1 lo
  g1[3] = (int)((tensor_rows >> 16) | ((tile_k & 0xFFFFu) << 16));   // dim1 hi | tile_dim0
  g1[4] = (int)(tile_rows & 0xFFFFu);            // tile_dim1 (tile_dim2 = 0)
  g1[5] = (int)stride_elems;                     // tensor_dim0_stride[31:0]
  g1[6] = 0;                                     // stride hi | dim1_stride lo
  g1[7] = 0;
  i32x4 z4 = {0, 0, 0, 0};
  i32x8 z8 = {0, 0, 0, 0, 0, 0, 0, 0};
  __builtin_amdgcn_tensor_load_to_lds(g0, g1, z4, z4, z8, 0);
}

// ---------------------------------------------------------------------------
// LayerNorm: one block (256 threads) per row; f32 in -> bf16 out
// ---------------------------------------------------------------------------
__global__ void layernorm_bf16_kernel(const float* __restrict__ x,
                                      const float* __restrict__ w,
                                      const float* __restrict__ b,
                                      __bf16* __restrict__ out, int C) {
  __shared__ float rbuf[256];
  size_t row = blockIdx.x;
  const float* xr = x + row * (size_t)C;
  int tid = threadIdx.x;
  float s = 0.f, ss = 0.f;
  for (int i = tid; i < C; i += 256) { float v = xr[i]; s += v; ss += v * v; }
  rbuf[tid] = s; __syncthreads();
  for (int off = 128; off; off >>= 1) { if (tid < off) rbuf[tid] += rbuf[tid + off]; __syncthreads(); }
  float mean = rbuf[0] / (float)C;
  __syncthreads();
  rbuf[tid] = ss; __syncthreads();
  for (int off = 128; off; off >>= 1) { if (tid < off) rbuf[tid] += rbuf[tid + off]; __syncthreads(); }
  float var = rbuf[0] / (float)C - mean * mean;
  float rstd = rsqrtf(var + 1e-5f);
  for (int i = tid; i < C; i += 256)
    out[row * (size_t)C + i] = to_bf16((xr[i] - mean) * rstd * w[i] + b[i]);
}

// ---------------------------------------------------------------------------
// Weight convert + transpose: f32 [batch,K,N] -> bf16 [batch,N,K]
// ---------------------------------------------------------------------------
__global__ void convert_wt_kernel(const float* __restrict__ W, __bf16* __restrict__ Wt,
                                  int batch, int K, int N) {
  size_t i = (size_t)blockIdx.x * 256 + threadIdx.x;
  size_t total = (size_t)batch * K * N;
  if (i >= total) return;
  size_t bh = i / ((size_t)K * N);
  size_t r  = i % ((size_t)K * N);
  size_t kk = r / N, nn = r % N;
  Wt[bh * (size_t)K * N + nn * (size_t)K + kk] = to_bf16(W[i]);
}

// ---------------------------------------------------------------------------
// QKV projection: out[b,h,s,d] (or transposed [b,h,d,s]) = X[row,:] @ W[h]
// Block: 128 threads (4 waves), each wave one 16-wide d tile; grid (M/16, H)
// ---------------------------------------------------------------------------
template <bool TRANS>
__global__ void qkv_proj_kernel(const __bf16* __restrict__ X,
                                const __bf16* __restrict__ Wt,   // [H, D, C]
                                __bf16* __restrict__ out) {
  int h = blockIdx.y;
  int wave = threadIdx.x >> 5, lane = threadIdx.x & 31;
  int row0 = blockIdx.x * 16;
  int n0 = wave * 16;
  const __bf16* WtH = Wt + (size_t)h * D_HEAD * C_DIM;
  f32x8 acc = {0.f, 0.f, 0.f, 0.f, 0.f, 0.f, 0.f, 0.f};
  for (int k0 = 0; k0 < C_DIM; k0 += 32) {
    bf16x16 a = load_a(X, C_DIM, row0, k0, lane);
    bf16x16 b = load_b(WtH, C_DIM, n0, k0, lane);
    acc = bf16_mma(a, b, acc);
  }
  int n = lane & 15, mh = lane >> 4;
#pragma unroll
  for (int r = 0; r < 8; ++r) {
    int row = row0 + r + 8 * mh;
    int bb = row / S_DIM, s = row % S_DIM;
    int d = n0 + n;
    if (TRANS)
      out[(((size_t)bb * H_DIM + h) * D_HEAD + d) * S_DIM + s] = to_bf16(acc[r]);
    else
      out[(((size_t)bb * H_DIM + h) * S_DIM + s) * D_HEAD + d] = to_bf16(acc[r]);
  }
}

// ---------------------------------------------------------------------------
// Attention: per (b,h, 16-row s tile). Full-row softmax, post-softmax strict
// upper-triangular mask, WMMA for QK^T and AV. 128 threads, dynamic LDS.
// ---------------------------------------------------------------------------
__global__ void attention_kernel(const __bf16* __restrict__ q,    // [B,H,S,64]
                                 const __bf16* __restrict__ kmat, // [B,H,S,64]
                                 const __bf16* __restrict__ vT,   // [B,H,64,S]
                                 __bf16* __restrict__ o) {        // [B*S, 1024]
  extern __shared__ char smem[];
  float*  sc  = (float*)smem;                                     // 16 x 2048 f32
  __bf16* pbf = (__bf16*)(smem + 16 * S_DIM * sizeof(float));     // 16 x 2048 bf16
  float*  red = (float*)(smem + 16 * S_DIM * (sizeof(float) + sizeof(__bf16)));

  int bh = blockIdx.y;                 // b*H + h
  int s0 = blockIdx.x * 16;
  const __bf16* qbh = q    + (size_t)bh * S_DIM * D_HEAD;
  const __bf16* kbh = kmat + (size_t)bh * S_DIM * D_HEAD;
  const __bf16* vbh = vT   + (size_t)bh * D_HEAD * S_DIM;
  int wave = threadIdx.x >> 5, lane = threadIdx.x & 31;
  int n = lane & 15, mh = lane >> 4;

  // --- Phase 1: scores = (Q K^T) * D^-0.5, full 16 x 2048 strip ---
  bf16x16 aq0 = load_a(qbh, D_HEAD, s0, 0, lane);
  bf16x16 aq1 = load_a(qbh, D_HEAD, s0, 32, lane);
  for (int tt = wave; tt < S_DIM / 16; tt += 4) {
    int t0 = tt * 16;
    f32x8 accs = {0.f, 0.f, 0.f, 0.f, 0.f, 0.f, 0.f, 0.f};
    accs = bf16_mma(aq0, load_b(kbh, D_HEAD, t0, 0, lane), accs);
    accs = bf16_mma(aq1, load_b(kbh, D_HEAD, t0, 32, lane), accs);
#pragma unroll
    for (int r = 0; r < 8; ++r)
      sc[(r + 8 * mh) * S_DIM + t0 + n] = accs[r] * 0.125f;  // 1/sqrt(64)
  }
  __syncthreads();

  // --- Phase 2: full-row softmax, then strict-upper mask, p -> bf16 LDS ---
  int row = threadIdx.x >> 3;          // 16 rows
  int seg = threadIdx.x & 7;           // 8 segments of 256
  float* srow = sc + row * S_DIM;
  int tb = seg * 256;
  float mx = -INFINITY;
  for (int i = 0; i < 256; ++i) mx = fmaxf(mx, srow[tb + i]);
  red[row * 8 + seg] = mx;
  __syncthreads();
  mx = red[row * 8];
  for (int j = 1; j < 8; ++j) mx = fmaxf(mx, red[row * 8 + j]);
  __syncthreads();
  float sum = 0.f;
  for (int i = 0; i < 256; ++i) sum += __expf(srow[tb + i] - mx);
  red[row * 8 + seg] = sum;
  __syncthreads();
  sum = 0.f;
  for (int j = 0; j < 8; ++j) sum += red[row * 8 + j];
  float inv = 1.f / sum;
  int sg = s0 + row;                   // global query index
  __bf16* prow = pbf + row * S_DIM;
  for (int i = 0; i < 256; ++i) {
    int t = tb + i;
    float p = __expf(srow[t] - mx) * inv;
    prow[t] = to_bf16((t > sg) ? p : 0.f);   // mask applied AFTER softmax
  }
  __syncthreads();

  // --- Phase 3: O = P @ V via WMMA, A from LDS, B from transposed V ---
  int n0 = wave * 16;
  f32x8 acco = {0.f, 0.f, 0.f, 0.f, 0.f, 0.f, 0.f, 0.f};
  for (int k0 = 0; k0 < S_DIM; k0 += 32) {
    bf16x16 ap = load_a(pbf, S_DIM, 0, k0, lane);
    bf16x16 bv = load_b(vbh, S_DIM, n0, k0, lane);
    acco = bf16_mma(ap, bv, acco);
  }
  int b = bh >> 4, h = bh & 15;
#pragma unroll
  for (int r = 0; r < 8; ++r) {
    int s = s0 + r + 8 * mh;
    o[((size_t)b * S_DIM + s) * (H_DIM * D_HEAD) + h * D_HEAD + n0 + n] = to_bf16(acco[r]);
  }
}

// ---------------------------------------------------------------------------
// TDM-staged, double-buffered tiled GEMM.
// Block 256 threads = 8 waves, tile 64(M) x 128(N); wave grid 2x4, each wave
// 32x32 output = 2x2 WMMA fragments (each LDS fragment feeds 2 WMMAs).
// K consumed in 32-wide chunks DMA'd into LDS by wave 0 via tensor_load_to_lds,
// synchronized with s_wait_tensorcnt + barriers.
// EPI 0: outf = res + acc + bias   (li1)
// EPI 1: outb = bf16(gelu(acc+b))  (li2)
// EPI 2: outf += acc + bias        (li3)
// ---------------------------------------------------------------------------
template <int EPI>
__global__ void gemm_bf16_tdm_kernel(const __bf16* __restrict__ A,   // [M,K]
                                     const __bf16* __restrict__ Bt,  // [N,K]
                                     const float* __restrict__ bias,
                                     const float* __restrict__ res,
                                     float* __restrict__ outf,
                                     __bf16* __restrict__ outb,
                                     int M, int N, int K) {
  __shared__ __bf16 smA[2][64 * 32];    // 4 KB per buffer
  __shared__ __bf16 smB[2][128 * 32];   // 8 KB per buffer
  int wave = threadIdx.x >> 5, lane = threadIdx.x & 31;
  int wm = wave >> 2, wn = wave & 3;
  int m0 = blockIdx.x * 64;
  int n0 = blockIdx.y * 128;
  int nk = K / 32;

  const __bf16* Atile = A + (size_t)m0 * (size_t)K;
  const __bf16* Btile = Bt + (size_t)n0 * (size_t)K;

  if (wave == 0) {   // prologue: stage k-chunk 0 into buffer 0
    tdm_load_2d((unsigned)(uintptr_t)&smA[0][0], Atile, 32, 64,
                (unsigned)K, (unsigned)M, (unsigned)K);
    tdm_load_2d((unsigned)(uintptr_t)&smB[0][0], Btile, 32, 128,
                (unsigned)K, (unsigned)N, (unsigned)K);
  }

  f32x8 zero = {0.f, 0.f, 0.f, 0.f, 0.f, 0.f, 0.f, 0.f};
  f32x8 acc00 = zero, acc01 = zero, acc10 = zero, acc11 = zero;

  for (int ks = 0; ks < nk; ++ks) {
    int cur = ks & 1, nxt = cur ^ 1;
    __syncthreads();   // all waves done reading buffer `nxt` (iteration ks-1)
    if (wave == 0) {
      if (ks + 1 < nk) {
        int k1 = (ks + 1) * 32;
        tdm_load_2d((unsigned)(uintptr_t)&smA[nxt][0], Atile + k1, 32, 64,
                    (unsigned)K, (unsigned)M, (unsigned)K);
        tdm_load_2d((unsigned)(uintptr_t)&smB[nxt][0], Btile + k1, 32, 128,
                    (unsigned)K, (unsigned)N, (unsigned)K);
        __builtin_amdgcn_s_wait_tensorcnt(2);  // the 2 DMAs for `cur` are done
      } else {
        __builtin_amdgcn_s_wait_tensorcnt(0);
      }
    }
    __syncthreads();   // buffer `cur` visible to all waves

    bf16x16 a0 = load_a(&smA[cur][0], 32, wm * 32 +  0, 0, lane);
    bf16x16 a1 = load_a(&smA[cur][0], 32, wm * 32 + 16, 0, lane);
    bf16x16 b0 = load_b(&smB[cur][0], 32, wn * 32 +  0, 0, lane);
    bf16x16 b1 = load_b(&smB[cur][0], 32, wn * 32 + 16, 0, lane);
    acc00 = bf16_mma(a0, b0, acc00);
    acc01 = bf16_mma(a0, b1, acc01);
    acc10 = bf16_mma(a1, b0, acc10);
    acc11 = bf16_mma(a1, b1, acc11);
  }

  int nl = lane & 15, mh = lane >> 4;
#pragma unroll
  for (int fm = 0; fm < 2; ++fm) {
#pragma unroll
    for (int fn = 0; fn < 2; ++fn) {
      f32x8 acc = (fm == 0) ? (fn == 0 ? acc00 : acc01)
                            : (fn == 0 ? acc10 : acc11);
      int col = n0 + wn * 32 + fn * 16 + nl;
      float bv = bias[col];
#pragma unroll
      for (int r = 0; r < 8; ++r) {
        int mrow = m0 + wm * 32 + fm * 16 + r + 8 * mh;
        size_t idx = (size_t)mrow * (size_t)N + col;
        float v = acc[r] + bv;
        if (EPI == 0)      outf[idx] = res[idx] + v;
        else if (EPI == 1) outb[idx] = to_bf16(gelu_exact(v));
        else               outf[idx] += v;
      }
    }
  }
}

// ---------------------------------------------------------------------------
// Host-side launcher
// ---------------------------------------------------------------------------
extern "C" void kernel_launch(void* const* d_in, const int* in_sizes, int n_in,
                              void* d_out, int out_size, void* d_ws, size_t ws_size,
                              hipStream_t stream) {
  const float* x     = (const float*)d_in[0];
  const float* y     = (const float*)d_in[1];
  const float* Wq    = (const float*)d_in[2];
  const float* Wk    = (const float*)d_in[3];
  const float* Wv    = (const float*)d_in[4];
  const float* li1_w = (const float*)d_in[5];
  const float* li1_b = (const float*)d_in[6];
  const float* ln1_w = (const float*)d_in[7];
  const float* ln1_b = (const float*)d_in[8];
  const float* ln2_w = (const float*)d_in[9];
  const float* ln2_b = (const float*)d_in[10];
  const float* ln3_w = (const float*)d_in[11];
  const float* ln3_b = (const float*)d_in[12];
  const float* li2_w = (const float*)d_in[13];
  const float* li2_b = (const float*)d_in[14];
  const float* li3_w = (const float*)d_in[15];
  const float* li3_b = (const float*)d_in[16];
  float* out = (float*)d_out;

  // ----- workspace carve (bf16 buffers) -----
  char* ws = (char*)d_ws;
  size_t off = 0;
  auto carve = [&](size_t bytes) -> void* {
    void* p = ws + off;
    off += (bytes + 255) & ~(size_t)255;
    return p;
  };
  const size_t MC = (size_t)M_ROWS * C_DIM;                 // 4096*1024
  __bf16* xn    = (__bf16*)carve(MC * 2);
  __bf16* yn    = (__bf16*)carve(MC * 2);
  __bf16* Wq_t  = (__bf16*)carve((size_t)H_DIM * D_HEAD * C_DIM * 2);
  __bf16* Wk_t  = (__bf16*)carve((size_t)H_DIM * D_HEAD * C_DIM * 2);
  __bf16* Wv_t  = (__bf16*)carve((size_t)H_DIM * D_HEAD * C_DIM * 2);
  __bf16* qbuf  = (__bf16*)carve(MC * 2);                   // [B,H,S,64]
  __bf16* kbuf  = (__bf16*)carve(MC * 2);                   // [B,H,S,64]
  __bf16* vTbuf = (__bf16*)carve(MC * 2);                   // [B,H,64,S]
  __bf16* obuf  = (__bf16*)carve(MC * 2);                   // [M, 1024]
  __bf16* w1t   = (__bf16*)carve((size_t)C_DIM * C_DIM * 2);
  __bf16* hbuf  = (__bf16*)carve(MC * 2);
  __bf16* w2t   = (__bf16*)carve((size_t)C_DIM * C_DIM * W_EXP * 2);
  __bf16* gbuf  = (__bf16*)carve((size_t)M_ROWS * C_DIM * W_EXP * 2);
  __bf16* w3t   = (__bf16*)carve((size_t)C_DIM * W_EXP * C_DIM * 2);
  (void)ws_size; (void)n_in; (void)in_sizes; (void)out_size;

  // ----- 1. LayerNorms on inputs -----
  layernorm_bf16_kernel<<<dim3(M_ROWS), dim3(256), 0, stream>>>(x, ln1_w, ln1_b, xn, C_DIM);
  layernorm_bf16_kernel<<<dim3(M_ROWS), dim3(256), 0, stream>>>(y, ln2_w, ln2_b, yn, C_DIM);

  // ----- 2. Weight conversions (f32 [.,K,N] -> bf16 [.,N,K]) -----
  auto cvt = [&](const float* W, __bf16* Wt, int batch, int K, int N) {
    size_t total = (size_t)batch * K * N;
    convert_wt_kernel<<<dim3((unsigned)((total + 255) / 256)), dim3(256), 0, stream>>>(W, Wt, batch, K, N);
  };
  cvt(Wq, Wq_t, H_DIM, C_DIM, D_HEAD);
  cvt(Wk, Wk_t, H_DIM, C_DIM, D_HEAD);
  cvt(Wv, Wv_t, H_DIM, C_DIM, D_HEAD);
  cvt(li1_w, w1t, 1, C_DIM, C_DIM);
  cvt(li2_w, w2t, 1, C_DIM, C_DIM * W_EXP);
  cvt(li3_w, w3t, 1, C_DIM * W_EXP, C_DIM);

  // ----- 3. Q/K/V projections (WMMA) -----
  dim3 qkv_grid(M_ROWS / 16, H_DIM);
  qkv_proj_kernel<false><<<qkv_grid, dim3(128), 0, stream>>>(xn, Wq_t, qbuf);
  qkv_proj_kernel<false><<<qkv_grid, dim3(128), 0, stream>>>(yn, Wk_t, kbuf);
  qkv_proj_kernel<true ><<<qkv_grid, dim3(128), 0, stream>>>(yn, Wv_t, vTbuf);

  // ----- 4. Attention (WMMA QK^T + softmax + post-mask + WMMA AV) -----
  size_t att_smem = (size_t)16 * S_DIM * sizeof(float)     // f32 scores
                  + (size_t)16 * S_DIM * sizeof(__bf16)    // bf16 probs
                  + (size_t)16 * 8 * sizeof(float);        // reductions
  attention_kernel<<<dim3(S_DIM / 16, B_DIM * H_DIM), dim3(128), att_smem, stream>>>(
      qbuf, kbuf, vTbuf, obuf);

  // ----- 5. li1: x_out = x + o @ li1_w + b  (f32 into d_out, TDM GEMM) -----
  gemm_bf16_tdm_kernel<0><<<dim3(M_ROWS / 64, C_DIM / 128), dim3(256), 0, stream>>>(
      obuf, w1t, li1_b, x, out, nullptr, M_ROWS, C_DIM, C_DIM);

  // ----- 6. ln3 on x_out -----
  layernorm_bf16_kernel<<<dim3(M_ROWS), dim3(256), 0, stream>>>(out, ln3_w, ln3_b, hbuf, C_DIM);

  // ----- 7. li2 + GELU (TDM GEMM) -----
  gemm_bf16_tdm_kernel<1><<<dim3(M_ROWS / 64, (C_DIM * W_EXP) / 128), dim3(256), 0, stream>>>(
      hbuf, w2t, li2_b, nullptr, nullptr, gbuf, M_ROWS, C_DIM * W_EXP, C_DIM);

  // ----- 8. li3 accumulated into d_out (final residual, TDM GEMM) -----
  gemm_bf16_tdm_kernel<2><<<dim3(M_ROWS / 64, C_DIM / 128), dim3(256), 0, stream>>>(
      gbuf, w3t, li3_b, nullptr, out, nullptr, M_ROWS, C_DIM, C_DIM * W_EXP);
}
